// simple_NN_5832565588325
// MI455X (gfx1250) — matrix operand order repaired and checked by reading, not compiled, not run on hardware
//
#include <hip/hip_runtime.h>

typedef __attribute__((ext_vector_type(16))) _Float16 v16h;
typedef __attribute__((ext_vector_type(8)))  float    v8f;

union BMat {
    v16h v;
    _Float16 h[16];
    unsigned u[8];
};

// hardware tanh (TRANS unit on gfx1250)
__device__ __forceinline__ float htanh(float x) {
#if __has_builtin(__builtin_amdgcn_tanhf)
    return __builtin_amdgcn_tanhf(x);
#elif __has_builtin(__builtin_amdgcn_tanh_f32)
    return __builtin_amdgcn_tanh_f32(x);
#else
    float r;
    asm("v_tanh_f32 %0, %1" : "=v"(r) : "v"(x));
    return r;
#endif
}

__device__ __forceinline__ unsigned packh2(float a, float b) {
    union { _Float16 h[2]; unsigned u; } p;
    p.h[0] = (_Float16)a;
    p.h[1] = (_Float16)b;
    return p.u;
}

// swap value between lane l and lane l^16 (SWAPX16: xor=0x10, and=0x1f)
__device__ __forceinline__ unsigned swap16(unsigned x) {
    return (unsigned)__builtin_amdgcn_ds_swizzle((int)x, 0x401F);
}

__device__ __forceinline__ v8f wmma16(v16h a, v16h b, v8f c) {
    // (neg_a, A, neg_b, B, c_mod, C, reuse_a, reuse_b)
    return __builtin_amdgcn_wmma_f32_16x16x32_f16(false, a, false, b, (short)0, c,
                                                  false, false);
}

// Build next-layer B (32 features) from two C tiles.
// C layout: lane-group g holds features 8g+r (lo) and 16+8g+r (hi) in VGPR r,
// batch column n = lane&15. B layout: lane-group g needs K=16g+i at half i.
// -> own tl covers K[0..7] (g=0) / own th covers K[24..31] (g=1); the other
//    8 features live in the partner lane -> one dword swap per pair.
__device__ __forceinline__ BMat make_B32(v8f c_lo, v8f c_hi, int g) {
    BMat B;
    unsigned pl[4], ph[4];
#pragma unroll
    for (int j = 0; j < 4; ++j) {
        pl[j] = packh2(htanh(c_lo[2 * j]), htanh(c_lo[2 * j + 1]));
        ph[j] = packh2(htanh(c_hi[2 * j]), htanh(c_hi[2 * j + 1]));
    }
#pragma unroll
    for (int j = 0; j < 4; ++j) {
        unsigned rv = swap16(g ? pl[j] : ph[j]);
        B.u[j]     = g ? rv : pl[j];
        B.u[4 + j] = g ? ph[j] : rv;
    }
    return B;
}

// Same for a 16-feature layer output (K=16 real, upper half of B is zero pad).
__device__ __forceinline__ BMat make_B16(v8f c, int g) {
    BMat B;
    unsigned pl[4];
#pragma unroll
    for (int j = 0; j < 4; ++j)
        pl[j] = packh2(htanh(c[2 * j]), htanh(c[2 * j + 1]));
#pragma unroll
    for (int j = 0; j < 4; ++j) {
        unsigned rv = swap16(pl[j]);
        B.u[j]     = g ? 0u : pl[j];   // g=1 lanes: K=16..31 -> zero pad
        B.u[4 + j] = g ? 0u : rv;
    }
    return B;
}

__global__ __launch_bounds__(256) void mlp5_wmma_kernel(
    const float* __restrict__ X,
    const float* __restrict__ W0, const float* __restrict__ b0,
    const float* __restrict__ W1, const float* __restrict__ b1,
    const float* __restrict__ W2, const float* __restrict__ b2,
    const float* __restrict__ W3, const float* __restrict__ b3,
    const float* __restrict__ W4, const float* __restrict__ b4,
    float* __restrict__ Out, int n)
{
    const int lane = (int)(threadIdx.x & 31);
    const int g    = lane >> 4;     // half-wave group
    const int nl   = lane & 15;     // batch column / A-row index
    const int koff = g * 8;

    // ---- A matrices: W^T, f16, per-lane row m = nl (+16 for "hi") ----
    // A k order used here (contiguous, same as B; not AMD's table order): lanes<16 hold K in {0..15},
    // lanes>=16 hold K in {16..31}; half index i -> K = i + 2*koff.
    BMat A1lo, A1hi, A2lo, A2hi, A3, A4;
#pragma unroll
    for (int i = 0; i < 16; ++i) {
        const int k = i + 2 * koff;
        A1lo.h[i] = (g == 0) ? (_Float16)W1[i * 32 + nl]        : (_Float16)0.f;
        A1hi.h[i] = (g == 0) ? (_Float16)W1[i * 32 + 16 + nl]   : (_Float16)0.f;
        A2lo.h[i] = (_Float16)W2[k * 32 + nl];
        A2hi.h[i] = (_Float16)W2[k * 32 + 16 + nl];
        A3.h[i]   = (_Float16)W3[k * 16 + nl];
        A4.h[i]   = (g == 0 && nl < 3) ? (_Float16)W4[i * 3 + nl]
                                      : (_Float16)0.f;
    }

    // ---- biases as WMMA C-inputs: feature m = 8g + r in VGPR r ----
    v8f vb1lo, vb1hi, vb2lo, vb2hi, vb3, vb4;
#pragma unroll
    for (int r = 0; r < 8; ++r) {
        vb1lo[r] = b1[koff + r];
        vb1hi[r] = b1[16 + koff + r];
        vb2lo[r] = b2[koff + r];
        vb2hi[r] = b2[16 + koff + r];
        vb3[r]   = b3[koff + r];
        vb4[r]   = (g == 0 && r < 3) ? b4[r] : 0.f;
    }

    // ---- layer-0 weights: wave-uniform -> scalar regs ----
    float w0v[16], b0v[16];
#pragma unroll
    for (int j = 0; j < 16; ++j) {
        w0v[j] = W0[j];
        b0v[j] = b0[j];
    }

    const int tiles  = (n + 15) >> 4;
    const int wave   = (int)((blockIdx.x * blockDim.x + threadIdx.x) >> 5);
    const int nwaves = (int)((gridDim.x * blockDim.x) >> 5);

    for (int t = wave; t < tiles; t += nwaves) {
        const int base = t << 4;
        const int row  = base + nl;
        const int ridx = (row < n) ? row : (n - 1);   // clamp, keep EXEC full
        const float x  = X[ridx];

        // L0: 1 -> 16 (VALU outer product). B0: g=0 lanes carry K=0..15,
        // g=1 lanes carry the K=16..31 zero padding.
        BMat B0;
#pragma unroll
        for (int j = 0; j < 8; ++j) {
            float e0 = htanh(fmaf(x, w0v[2 * j],     b0v[2 * j]));
            float e1 = htanh(fmaf(x, w0v[2 * j + 1], b0v[2 * j + 1]));
            B0.u[j] = (g == 0) ? packh2(e0, e1) : 0u;
        }

        // L1: 16 -> 32
        v8f c_lo = wmma16(A1lo.v, B0.v, vb1lo);
        v8f c_hi = wmma16(A1hi.v, B0.v, vb1hi);
        BMat B1 = make_B32(c_lo, c_hi, g);

        // L2: 32 -> 32
        c_lo = wmma16(A2lo.v, B1.v, vb2lo);
        c_hi = wmma16(A2hi.v, B1.v, vb2hi);
        BMat B2 = make_B32(c_lo, c_hi, g);

        // L3: 32 -> 16
        v8f c3 = wmma16(A3.v, B2.v, vb3);
        BMat B3 = make_B16(c3, g);

        // L4: 16 -> 3 (rows 3..15 of A are zero)
        v8f c4 = wmma16(A4.v, B3.v, vb4);

        // Output: g=0 lanes hold features 0..7 of batch column nl -> store 0..2
        if (g == 0 && row < n) {
            float* o = Out + (size_t)row * 3;
            o[0] = htanh(c4[0]);
            o[1] = htanh(c4[1]);
            o[2] = htanh(c4[2]);
        }
    }
}

extern "C" void kernel_launch(void* const* d_in, const int* in_sizes, int n_in,
                              void* d_out, int out_size, void* d_ws, size_t ws_size,
                              hipStream_t stream) {
    (void)n_in; (void)out_size; (void)d_ws; (void)ws_size;
    const float* X  = (const float*)d_in[0];
    const float* W0 = (const float*)d_in[1];
    const float* b0 = (const float*)d_in[2];
    const float* W1 = (const float*)d_in[3];
    const float* b1 = (const float*)d_in[4];
    const float* W2 = (const float*)d_in[5];
    const float* b2 = (const float*)d_in[6];
    const float* W3 = (const float*)d_in[7];
    const float* b3 = (const float*)d_in[8];
    const float* W4 = (const float*)d_in[9];
    const float* b4 = (const float*)d_in[10];

    const int n = in_sizes[0];
    const int threads = 256;
    const int wavesPerBlock = threads / 32;
    const int tiles = (n + 15) / 16;
    int blocks = (tiles + wavesPerBlock - 1) / wavesPerBlock;
    if (blocks > 2048) blocks = 2048;
    if (blocks < 1) blocks = 1;

    mlp5_wmma_kernel<<<blocks, threads, 0, stream>>>(
        X, W0, b0, W1, b1, W2, b2, W3, b3, W4, b4, (float*)d_out, n);
}
